// PolarNet_74947179315709
// MI455X (gfx1250) — matrix operand order repaired and checked by reading, not compiled, not run on hardware
//
#include <hip/hip_runtime.h>
#include <hip/hip_bf16.h>

// ---------------------------------------------------------------------------
// PolarNet fused pipeline for gfx1250 (MI455X)
//   B=16, C=256, H=W=32, HW=1024
//   Stage 1: QKV 1x1-conv as bf16 WMMA GEMM (v_wmma_f32_16x16x32_bf16)
//            with async global->LDS double-buffered staging (ASYNCcnt)
//   Stage 2: 9-neighbor attention (wave32 shfl reductions)
//   Stage 3: residual + LayerNorm over HW
// ---------------------------------------------------------------------------

#define Bn   16
#define Cn   256
#define HWn  1024
#define Hn   32
#define Wn   32

typedef __attribute__((ext_vector_type(16))) __bf16          v16bf;
typedef __attribute__((ext_vector_type(8)))  __bf16          v8bf;
typedef __attribute__((ext_vector_type(8)))  float           v8f;
typedef __attribute__((ext_vector_type(8)))  unsigned short  v8us;
typedef __attribute__((ext_vector_type(4)))  int             v4i;

// LDS row strides (halves) chosen so 16-lane fragment reads are bank-conflict
// free: XS 40 halves (80B -> bank step 20, gcd(20,64)=4, 16 distinct banks);
// WS 264 halves (528B -> bank step 4, 16 distinct banks).
#define XS_STRIDE 40
#define WS_STRIDE 264

#if defined(__has_builtin)
# if __has_builtin(__builtin_amdgcn_global_load_async_to_lds_b128)
#  define HAVE_ASYNC_LDS 1
# endif
#endif

typedef __attribute__((address_space(1))) v4i as1_v4i;
typedef __attribute__((address_space(3))) v4i as3_v4i;

// 16-byte global -> LDS copy (async DMA when available)
__device__ __forceinline__ void async_cp16(const unsigned short* g, unsigned short* l) {
#ifdef HAVE_ASYNC_LDS
    __builtin_amdgcn_global_load_async_to_lds_b128((as1_v4i*)g, (as3_v4i*)l, 0, 0);
#else
    *(v8us*)l = *(const v8us*)g;
#endif
}

__device__ __forceinline__ void wait_async0() {
#ifdef HAVE_ASYNC_LDS
# if __has_builtin(__builtin_amdgcn_s_wait_asynccnt)
    __builtin_amdgcn_s_wait_asynccnt(0);
# else
    asm volatile("s_wait_asynccnt 0x0" ::: "memory");
# endif
#endif
}

// round-to-nearest-even f32 -> bf16 (bit pattern as ushort)
__device__ __forceinline__ unsigned short f2bf(float f) {
    unsigned u = __builtin_bit_cast(unsigned, f);
    unsigned r = u + 0x7FFFu + ((u >> 16) & 1u);
    return (unsigned short)(r >> 16);
}

// ---------------------------------------------------------------------------
// Kernel 1: weights f32 -> bf16 (row-major o x c, unchanged layout)
// ---------------------------------------------------------------------------
__global__ void prep_w(const float* __restrict__ w1, const float* __restrict__ w2,
                       const float* __restrict__ w3,
                       unsigned short* __restrict__ wq, unsigned short* __restrict__ wk,
                       unsigned short* __restrict__ wv) {
    int i = blockIdx.x * blockDim.x + threadIdx.x;   // 0 .. 65535
    wq[i] = f2bf(w1[i]);
    wk[i] = f2bf(w2[i]);
    wv[i] = f2bf(w3[i]);
}

// ---------------------------------------------------------------------------
// Kernel 2: x (b,c,n) f32 -> xt (b,n,c) bf16, LDS-tiled transpose (coalesced
// on both sides).  Grid: (HW/32, C/32, B), block (32,8).
// ---------------------------------------------------------------------------
__global__ void prep_xt(const float* __restrict__ x, unsigned short* __restrict__ xt) {
    __shared__ float tile[32][33];
    const int b  = blockIdx.z;
    const int n0 = blockIdx.x * 32;
    const int c0 = blockIdx.y * 32;
    const int tx = threadIdx.x, ty = threadIdx.y;
#pragma unroll
    for (int i = 0; i < 4; ++i) {
        int c = c0 + ty + i * 8;
        tile[ty + i * 8][tx] = x[((long)b * Cn + c) * HWn + n0 + tx];
    }
    __syncthreads();
#pragma unroll
    for (int i = 0; i < 4; ++i) {
        int n = ty + i * 8;
        xt[((long)b * HWn + n0 + n) * Cn + c0 + tx] = f2bf(tile[tx][n]);
    }
}

// ---------------------------------------------------------------------------
// WMMA fragment builders from LDS (layouts per CDNA5 ISA 7.12.2, wave32)
// A (16x32 bf16, M x K): lane holds row M = lane&15;
//   halves 0..7  -> K = k0 + 8*(lane>>4) + i
//   halves 8..15 -> K = k0 + 8*(lane>>4) + 16 + (i-8)
// B (32x16 bf16, K x N): lane holds col N = lane&15;
//   halves 0..7  -> K = k0 + 8*(lane>>4) + i;  halves 8..15 -> K = k0 + 8*(lane>>4) + 16 + (i-8)
// ---------------------------------------------------------------------------
__device__ __forceinline__ v16bf lds_afrag(const unsigned short* ws, int k0, int lane) {
    const unsigned short* p = ws + (lane & 15) * WS_STRIDE + k0 + 8 * (lane >> 4);
    v8bf lo = __builtin_bit_cast(v8bf, *(const v8us*)p);
    v8bf hi = __builtin_bit_cast(v8bf, *(const v8us*)(p + 16));
    return __builtin_shufflevector(lo, hi, 0,1,2,3,4,5,6,7,8,9,10,11,12,13,14,15);
}

__device__ __forceinline__ v16bf lds_bfrag(const unsigned short* xs, int wave, int lane) {
    const unsigned short* p = xs + (wave * 16 + (lane & 15)) * XS_STRIDE + 8 * (lane >> 4);
    v8bf lo = __builtin_bit_cast(v8bf, *(const v8us*)p);
    v8bf hi = __builtin_bit_cast(v8bf, *(const v8us*)(p + 16));
    return __builtin_shufflevector(lo, hi, 0,1,2,3,4,5,6,7,8,9,10,11,12,13,14,15);
}

// ---------------------------------------------------------------------------
// Kernel 3: QKV GEMM.  D[o,n] = W[o,:] . X[:,n] + bias[o]
// Block = 8 waves, one 16(M) x 128(N) macro-tile, K=256 in 8 steps of 32.
//   * 3 weight tiles (16x256 bf16 each) staged to LDS once per block
//   * X K-slabs (128 pixels x 32 K) double-buffered via async global->LDS DMA
//   * each wave: one 16x16 N-subtile, 3 accumulators (Q,K,V) sharing B frag
// Outputs stored pixel-major f32: qt/kt/vt [b][n][c].
// ---------------------------------------------------------------------------
__global__ void __launch_bounds__(256) qkv_gemm(
        const unsigned short* __restrict__ wq, const unsigned short* __restrict__ wk,
        const unsigned short* __restrict__ wv, const unsigned short* __restrict__ xt,
        const float* __restrict__ bq, const float* __restrict__ bk,
        const float* __restrict__ bv,
        float* __restrict__ qt, float* __restrict__ kt, float* __restrict__ vt) {
    __shared__ unsigned short lds_w[3][16 * WS_STRIDE];   // 25344 B
    __shared__ unsigned short lds_x[2][128 * XS_STRIDE];  // 20480 B

    const int t    = threadIdx.x;
    const int lane = t & 31;
    const int wave = t >> 5;                       // 0..7
    const int b    = blockIdx.z;
    const int m0   = blockIdx.y * 16;              // output-channel tile
    const int n0   = blockIdx.x * 128;             // pixel macro-tile
    const unsigned short* xb = xt + ((long)b * HWn + n0) * Cn;

    // ---- stage the three 16x256 weight tiles (once) -------------------------
    {
        const int row  = t >> 4;            // 0..15
        const int hoff = (t & 15) * 16;     // 16 halves per thread per matrix
        const unsigned short* srcs[3] = { wq, wk, wv };
#pragma unroll
        for (int m = 0; m < 3; ++m) {
            const unsigned short* g = srcs[m] + (m0 + row) * Cn + hoff;
            unsigned short*       l = &lds_w[m][row * WS_STRIDE + hoff];
            async_cp16(g, l);
            async_cp16(g + 8, l + 8);
        }
    }
    // ---- stage X slab for K-step 0 -----------------------------------------
    {
        const int p  = t >> 1;
        const int co = (t & 1) * 16;
        const unsigned short* g = xb + (long)p * Cn + co;
        unsigned short*       l = &lds_x[0][p * XS_STRIDE + co];
        async_cp16(g, l);
        async_cp16(g + 8, l + 8);
    }
    wait_async0();
    __syncthreads();

    v8f accq = {}; v8f acck = {}; v8f accv = {};
#pragma unroll
    for (int ks = 0; ks < 8; ++ks) {
        // issue DMA for next K-slab while computing on the current one
        if (ks < 7) {
            const int p  = t >> 1;
            const int co = (t & 1) * 16;
            const unsigned short* g = xb + (long)p * Cn + (ks + 1) * 32 + co;
            unsigned short*       l = &lds_x[(ks + 1) & 1][p * XS_STRIDE + co];
            async_cp16(g, l);
            async_cp16(g + 8, l + 8);
        }
        const int k0 = ks * 32;
        v16bf bf = lds_bfrag(&lds_x[ks & 1][0], wave, lane);
        v16bf aq = lds_afrag(&lds_w[0][0], k0, lane);
        v16bf ak = lds_afrag(&lds_w[1][0], k0, lane);
        v16bf av = lds_afrag(&lds_w[2][0], k0, lane);
        accq = __builtin_amdgcn_wmma_f32_16x16x32_bf16(false, aq, false, bf, (short)0, accq, false, false);
        acck = __builtin_amdgcn_wmma_f32_16x16x32_bf16(false, ak, false, bf, (short)0, acck, false, false);
        accv = __builtin_amdgcn_wmma_f32_16x16x32_bf16(false, av, false, bf, (short)0, accv, false, false);
        if (ks < 7) {
            wait_async0();       // next slab landed
            __syncthreads();     // safe to read it / overwrite the other one
        }
    }

    // Epilogue: C/D layout -> lane holds N = lane&15, M = r + 8*(lane>>4)
    const int n     = n0 + wave * 16 + (lane & 15);
    const int cbase = m0 + 8 * (lane >> 4);
    const long ob   = ((long)b * HWn + n) * Cn + cbase;
#pragma unroll
    for (int r = 0; r < 8; ++r) {
        qt[ob + r] = accq[r] + bq[cbase + r];
        kt[ob + r] = acck[r] + bk[cbase + r];
        vt[ob + r] = accv[r] + bv[cbase + r];
    }
}

// ---------------------------------------------------------------------------
// Kernel 4: 9-neighbor attention.  Wave per pixel; lane owns channels
// c = lane + 32*t (t=0..7) so all q/k/v accesses are coalesced.
// Writes: new_v [b][n][c] f32, polarity, 2-channel output, softmax scores.
// ---------------------------------------------------------------------------
__global__ void attn(const float* __restrict__ qt, const float* __restrict__ kt,
                     const float* __restrict__ vt, float* __restrict__ newv,
                     float* __restrict__ pol, float* __restrict__ out2,
                     float* __restrict__ scores) {
    const int lane = threadIdx.x & 31;
    const int wid  = blockIdx.x * (blockDim.x >> 5) + (threadIdx.x >> 5);
    const int b = wid >> 10;
    const int n = wid & (HWn - 1);
    const int h = n >> 5, w = n & 31;
    const long base = (long)b * HWn * Cn;

    float q[8];
    const float* qp = qt + base + (long)n * Cn;
#pragma unroll
    for (int t = 0; t < 8; ++t) q[t] = qp[lane + 32 * t];

    float att[9];
    float nv[8];
#pragma unroll
    for (int t = 0; t < 8; ++t) nv[t] = 0.f;

#pragma unroll
    for (int j = 0; j < 9; ++j) {
        const int di = j / 3 - 1, dj = j % 3 - 1;
        int hh = h + di; hh = hh < 0 ? 0 : (hh > Hn - 1 ? Hn - 1 : hh);
        int ww = w + dj; ww = ww < 0 ? 0 : (ww > Wn - 1 ? Wn - 1 : ww);
        const int m = hh * Wn + ww;
        const float* kp = kt + base + (long)m * Cn;
        const float* vp = vt + base + (long)m * Cn;
        float s = 0.f;
#pragma unroll
        for (int t = 0; t < 8; ++t) s += q[t] * kp[lane + 32 * t];
#pragma unroll
        for (int off = 16; off > 0; off >>= 1) s += __shfl_xor(s, off, 32);
        att[j] = s;
#pragma unroll
        for (int t = 0; t < 8; ++t) nv[t] += s * vp[lane + 32 * t];
    }

    float* np = newv + base + (long)n * Cn;
#pragma unroll
    for (int t = 0; t < 8; ++t) np[lane + 32 * t] = nv[t];

    if (lane == 0) {
        float mx = att[0]; int am = 0;
#pragma unroll
        for (int j = 1; j < 9; ++j) if (att[j] > mx) { mx = att[j]; am = j; }
        float e[9], den = 0.f;
#pragma unroll
        for (int j = 0; j < 9; ++j) { e[j] = __expf(att[j] - mx); den += e[j]; }
        const float inv = 1.f / den;
        float* sp = scores + ((long)b * HWn + n) * 9;
#pragma unroll
        for (int j = 0; j < 9; ++j) sp[j] = e[j] * inv;
        pol[b * HWn + n] = (float)am;
        float oth = (att[0] + att[1] + att[2] + att[3] +
                     att[5] + att[6] + att[7] + att[8]) * 0.125f;
        out2[((long)b * 2 + 0) * HWn + n] = att[4];
        out2[((long)b * 2 + 1) * HWn + n] = oth;
    }
}

// ---------------------------------------------------------------------------
// Kernel 5: residual + LayerNorm over HW per (b,c).
// Block per (b,c); 256 threads, 4 pixels each.  x reads and xn writes are
// coalesced; new_v is a stride-C gather (hot in L2).
// ---------------------------------------------------------------------------
__global__ void lnorm(const float* __restrict__ x, const float* __restrict__ newv,
                      float* __restrict__ xn) {
    const int bc = blockIdx.x;           // b*256 + c
    const int b  = bc >> 8;
    const int c  = bc & 255;
    const float* xp  = x + (long)bc * HWn;
    const float* nvb = newv + (long)b * HWn * Cn + c;

    float vals[4], s = 0.f, s2 = 0.f;
#pragma unroll
    for (int i = 0; i < 4; ++i) {
        const int n = threadIdx.x + i * 256;
        const float v = xp[n] + nvb[(long)n * Cn];
        vals[i] = v; s += v; s2 += v * v;
    }
#pragma unroll
    for (int off = 16; off > 0; off >>= 1) {
        s  += __shfl_xor(s,  off, 32);
        s2 += __shfl_xor(s2, off, 32);
    }
    __shared__ float ss[8], ss2[8];
    const int lane = threadIdx.x & 31, wv = threadIdx.x >> 5;
    if (lane == 0) { ss[wv] = s; ss2[wv] = s2; }
    __syncthreads();
    if (wv == 0) {
        float a  = lane < 8 ? ss[lane]  : 0.f;
        float a2 = lane < 8 ? ss2[lane] : 0.f;
#pragma unroll
        for (int off = 4; off > 0; off >>= 1) {
            a  += __shfl_xor(a,  off, 32);
            a2 += __shfl_xor(a2, off, 32);
        }
        if (lane == 0) { ss[0] = a; ss2[0] = a2; }
    }
    __syncthreads();
    const float mean = ss[0] * (1.f / HWn);
    const float var  = ss2[0] * (1.f / HWn) - mean * mean;
    const float rstd = rsqrtf(var + 1e-5f);
#pragma unroll
    for (int i = 0; i < 4; ++i)
        xn[(long)bc * HWn + threadIdx.x + i * 256] = (vals[i] - mean) * rstd;
}

// ---------------------------------------------------------------------------
// Launch
// ---------------------------------------------------------------------------
extern "C" void kernel_launch(void* const* d_in, const int* in_sizes, int n_in,
                              void* d_out, int out_size, void* d_ws, size_t ws_size,
                              hipStream_t stream) {
    const float* x  = (const float*)d_in[0];
    const float* w1 = (const float*)d_in[1];
    const float* b1 = (const float*)d_in[2];
    const float* w2 = (const float*)d_in[3];
    const float* b2 = (const float*)d_in[4];
    const float* w3 = (const float*)d_in[5];
    const float* b3 = (const float*)d_in[6];

    // workspace layout (bytes)
    char* ws = (char*)d_ws;
    unsigned short* wq = (unsigned short*)(ws);                       // 64K bf16
    unsigned short* wk = (unsigned short*)(ws + 131072);
    unsigned short* wv = (unsigned short*)(ws + 262144);
    unsigned short* xt = (unsigned short*)(ws + 393216);              // 4M bf16
    float* qt   = (float*)(ws + 393216 + 8388608);                    // 4M f32 each
    float* kt   = qt + (long)Bn * HWn * Cn;
    float* vt   = kt + (long)Bn * HWn * Cn;
    float* newv = vt + (long)Bn * HWn * Cn;

    // d_out layout: polarity | output(2ch) | scores | xn
    float* pol    = (float*)d_out;                                    // 16384
    float* out2   = pol + Bn * HWn;                                   // 32768
    float* scores = out2 + Bn * 2 * HWn;                              // 147456
    float* xn     = scores + Bn * HWn * 9;                            // 4194304

    prep_w <<<256, 256, 0, stream>>>(w1, w2, w3, wq, wk, wv);
    prep_xt<<<dim3(HWn / 32, Cn / 32, Bn), dim3(32, 8), 0, stream>>>(x, xt);
    qkv_gemm<<<dim3(HWn / 128, Cn / 16, Bn), 256, 0, stream>>>(
        wq, wk, wv, xt, b1, b2, b3, qt, kt, vt);
    attn  <<<Bn * HWn / 8, 256, 0, stream>>>(qt, kt, vt, newv, pol, out2, scores);
    lnorm <<<Bn * Cn, 256, 0, stream>>>(x, newv, xn);
}
